// FA_37538014167570
// MI455X (gfx1250) — matrix operand order repaired and checked by reading, not compiled, not run on hardware
//
#include <hip/hip_runtime.h>
#include <math.h>

typedef __attribute__((ext_vector_type(16))) _Float16 v16h;
typedef __attribute__((ext_vector_type(8)))  float    v8f;

#define NCHAN   64
#define QKC     32
#define HW      65536
#define NBATCH  16
#define CHW     (NCHAN * HW)          // 4194304 floats per batch
#define BN_EPS  1e-5f
#define NRM_EPS 1e-12f

// ---- workspace layout (bytes) ----
// [0, 131072)        : f buffer  f32  [16][32][64]
// [131072, 132096)   : folded biases f32: bq(32) bk(32) bv(64) bo(64)
// [132096, 156672)   : f16 weight A-fragments: wq(2048) wk(2048) wv(4096) wo(4096) halves
#define F_OFF   0
#define B_OFF   131072
#define WF_OFF  132096

union FragCvt { v16h h16; unsigned int u[8]; };

// A-fragment (16x32 f16) element->K mapping (contiguous, matches B):
// lanes0-15 K=0..15, lanes16-31 K=16..31
__device__ __forceinline__ int kA_of(int lane, int h) {
  return ((lane >= 16) ? 16 : 0) + h;
}
// B-fragment (32x16 f16): VGPR j: lanes0-15 K=2j,2j+1; lanes16-31 K=16+2j,16+2j+1
// => element h: K = h + (hi?16:0)   (pairs are K-contiguous -> packed dword LDS loads)

__device__ __forceinline__ v8f wmma16(v16h a, v16h b, v8f c) {
  return __builtin_amdgcn_wmma_f32_16x16x32_f16(false, a, false, b, (short)0, c, false, false);
}

// ---------------- prep: fold BN, build f16 A-fragment weight images ----------------
__global__ __launch_bounds__(256) void fmav_prep(
    const float* __restrict__ Wq, const float* __restrict__ bq,
    const float* __restrict__ qg, const float* __restrict__ qb,
    const float* __restrict__ qm, const float* __restrict__ qv,
    const float* __restrict__ Wk, const float* __restrict__ bk,
    const float* __restrict__ kg, const float* __restrict__ kb,
    const float* __restrict__ km, const float* __restrict__ kv,
    const float* __restrict__ Wv, const float* __restrict__ bv,
    const float* __restrict__ vg, const float* __restrict__ vb,
    const float* __restrict__ vm, const float* __restrict__ vv,
    const float* __restrict__ Wo, const float* __restrict__ bo,
    const float* __restrict__ og, const float* __restrict__ ob,
    const float* __restrict__ om, const float* __restrict__ ov,
    float* __restrict__ biasOut, _Float16* __restrict__ wfOut)
{
  const int t = threadIdx.x;
  if (t < 32) {
    float s = qg[t] * rsqrtf(qv[t] + BN_EPS);
    biasOut[t] = bq[t] * s + (qb[t] - qm[t] * s);
    float sk = kg[t] * rsqrtf(kv[t] + BN_EPS);
    biasOut[32 + t] = bk[t] * sk + (kb[t] - km[t] * sk);
  }
  if (t < 64) {
    float s = vg[t] * rsqrtf(vv[t] + BN_EPS);
    biasOut[64 + t] = bv[t] * s + (vb[t] - vm[t] * s);
    float so = og[t] * rsqrtf(ov[t] + BN_EPS);
    biasOut[128 + t] = bo[t] * so + (ob[t] - om[t] * so);
  }
  // wq: 2 mb x 2 ks fragments (2048 halves)
  for (int idx = t; idx < 2048; idx += 256) {
    int frag = idx >> 9, lane = (idx >> 4) & 31, h = idx & 15;
    int mb = frag >> 1, ks = frag & 1;
    int row = mb * 16 + (lane & 15);
    int k = ks * 32 + kA_of(lane, h);
    float s = qg[row] * rsqrtf(qv[row] + BN_EPS);
    wfOut[idx] = (_Float16)(Wq[row * 64 + k] * s);
  }
  // wk (2048 halves at +2048)
  for (int idx = t; idx < 2048; idx += 256) {
    int frag = idx >> 9, lane = (idx >> 4) & 31, h = idx & 15;
    int mb = frag >> 1, ks = frag & 1;
    int row = mb * 16 + (lane & 15);
    int k = ks * 32 + kA_of(lane, h);
    float s = kg[row] * rsqrtf(kv[row] + BN_EPS);
    wfOut[2048 + idx] = (_Float16)(Wk[row * 64 + k] * s);
  }
  // wv: 4 mb x 2 ks (4096 halves at +4096)
  for (int idx = t; idx < 4096; idx += 256) {
    int frag = idx >> 9, lane = (idx >> 4) & 31, h = idx & 15;
    int mb = frag >> 1, ks = frag & 1;
    int row = mb * 16 + (lane & 15);
    int k = ks * 32 + kA_of(lane, h);
    float s = vg[row] * rsqrtf(vv[row] + BN_EPS);
    wfOut[4096 + idx] = (_Float16)(Wv[row * 64 + k] * s);
  }
  // wo: 4 mb x 2 ks (4096 halves at +8192)
  for (int idx = t; idx < 4096; idx += 256) {
    int frag = idx >> 9, lane = (idx >> 4) & 31, h = idx & 15;
    int mb = frag >> 1, ks = frag & 1;
    int row = mb * 16 + (lane & 15);
    int k = ks * 32 + kA_of(lane, h);
    float s = og[row] * rsqrtf(ov[row] + BN_EPS);
    wfOut[8192 + idx] = (_Float16)(Wo[row * 64 + k] * s);
  }
}

// ---------------- pass 1: f[n] = sum_s Kn[:,s] x relu(V)[s,:] ----------------
__global__ __launch_bounds__(256) void fmav_pass1(
    const float* __restrict__ x, const _Float16* __restrict__ wf,
    const float* __restrict__ bias, float* __restrict__ fbuf)
{
  const int n    = blockIdx.y;
  const int wid  = threadIdx.x >> 5;
  const int lane = threadIdx.x & 31;
  const int li   = lane & 15;
  const int hi   = (lane >> 4) & 1;

  __shared__ float    fAcc[32 * 64];
  __shared__ _Float16 ldsK[8][32 * 34];
  __shared__ _Float16 ldsV[8][64 * 34];

  for (int i = threadIdx.x; i < 2048; i += 256) fAcc[i] = 0.0f;
  __syncthreads();

  const float*     xn  = x + (size_t)n * CHW;
  const _Float16*  wkA = wf + 2048;
  const _Float16*  wvA = wf + 4096;
  const float*     bkF = bias + 32;
  const float*     bvF = bias + 64;

  v8f cbK[2], cbV[4];
  #pragma unroll
  for (int mb = 0; mb < 2; ++mb)
    #pragma unroll
    for (int r = 0; r < 8; ++r) cbK[mb][r] = bkF[mb * 16 + r + hi * 8];
  #pragma unroll
  for (int mb = 0; mb < 4; ++mb)
    #pragma unroll
    for (int r = 0; r < 8; ++r) cbV[mb][r] = bvF[mb * 16 + r + hi * 8];

  v8f facc[2][4];
  #pragma unroll
  for (int mb = 0; mb < 2; ++mb)
    #pragma unroll
    for (int nb = 0; nb < 4; ++nb)
      #pragma unroll
      for (int r = 0; r < 8; ++r) facc[mb][nb][r] = 0.0f;

  const int pBase = (blockIdx.x * 8 + wid) * 128;   // 64 blocks x 8 waves cover HW, 4 iters x 32 pos
  _Float16* lk = ldsK[wid];
  _Float16* lv = ldsV[wid];

  for (int it = 0; it < 4; ++it) {
    const int p0 = pBase + it * 32;
    if (it < 3) __builtin_prefetch(xn + p0 + 32 + (size_t)(hi * 16) * HW, 0, 0);

    #pragma unroll
    for (int t = 0; t < 2; ++t) {
      const int pt = p0 + t * 16;
      const float* xp = xn + pt;
      FragCvt bx0, bx1;
      #pragma unroll
      for (int h = 0; h < 16; ++h) {
        int cA = h + hi * 16;
        bx0.h16[h] = (_Float16)xp[(size_t)cA * HW + li];
        bx1.h16[h] = (_Float16)xp[(size_t)(cA + 32) * HW + li];
      }
      // ---- K projection (2 M-blocks), bias via C operand ----
      v8f kd[2];
      #pragma unroll
      for (int mb = 0; mb < 2; ++mb) {
        v16h a0 = *(const v16h*)(wkA + (mb * 2 + 0) * 512 + lane * 16);
        v16h a1 = *(const v16h*)(wkA + (mb * 2 + 1) * 512 + lane * 16);
        v8f acc = cbK[mb];
        acc = wmma16(a0, bx0.h16, acc);
        acc = wmma16(a1, bx1.h16, acc);
        kd[mb] = acc;
      }
      // per-position L2 norm over the 32 K-channels (column reduce across lane halves)
      float ss = 0.0f;
      #pragma unroll
      for (int mb = 0; mb < 2; ++mb)
        #pragma unroll
        for (int r = 0; r < 8; ++r) ss += kd[mb][r] * kd[mb][r];
      ss += __shfl_xor(ss, 16, 32);
      float rn = 1.0f / fmaxf(sqrtf(ss), NRM_EPS);
      #pragma unroll
      for (int mb = 0; mb < 2; ++mb)
        #pragma unroll
        for (int r = 0; r < 8; ++r)
          lk[(mb * 16 + r + hi * 8) * 34 + t * 16 + li] = (_Float16)(kd[mb][r] * rn);
      // ---- V projection (4 M-blocks) + ReLU ----
      #pragma unroll
      for (int mb = 0; mb < 4; ++mb) {
        v16h a0 = *(const v16h*)(wvA + (mb * 2 + 0) * 512 + lane * 16);
        v16h a1 = *(const v16h*)(wvA + (mb * 2 + 1) * 512 + lane * 16);
        v8f acc = cbV[mb];
        acc = wmma16(a0, bx0.h16, acc);
        acc = wmma16(a1, bx1.h16, acc);
        #pragma unroll
        for (int r = 0; r < 8; ++r)
          lv[(mb * 16 + r + hi * 8) * 34 + t * 16 + li] = (_Float16)fmaxf(acc[r], 0.0f);
      }
    }
    // ---- f += Kn[32ch x 32pos] * V^T[32pos x 64ch]  (K-dim = 32 positions) ----
    v16h aK[2];
    #pragma unroll
    for (int mb = 0; mb < 2; ++mb) {
      FragCvt fc;
      int row = mb * 16 + li;
      #pragma unroll
      for (int i = 0; i < 8; ++i) {
        int k0 = hi * 16 + 2 * i;     // kA_of(lane,2i)
        fc.u[i] = *(const unsigned int*)&lk[row * 34 + k0];
      }
      aK[mb] = fc.h16;
    }
    v16h bV[4];
    #pragma unroll
    for (int nb = 0; nb < 4; ++nb) {
      FragCvt fc;
      int row = nb * 16 + li;
      #pragma unroll
      for (int i = 0; i < 8; ++i) {
        int k0 = 2 * i + hi * 16;                               // kB(lane,2i)
        fc.u[i] = *(const unsigned int*)&lv[row * 34 + k0];
      }
      bV[nb] = fc.h16;
    }
    #pragma unroll
    for (int mb = 0; mb < 2; ++mb)
      #pragma unroll
      for (int nb = 0; nb < 4; ++nb)
        facc[mb][nb] = wmma16(aK[mb], bV[nb], facc[mb][nb]);
  }

  // block-level LDS reduce, then global float atomics into f[n]
  #pragma unroll
  for (int mb = 0; mb < 2; ++mb)
    #pragma unroll
    for (int nb = 0; nb < 4; ++nb)
      #pragma unroll
      for (int r = 0; r < 8; ++r)
        atomicAdd(&fAcc[(mb * 16 + r + hi * 8) * 64 + nb * 16 + li], facc[mb][nb][r]);
  __syncthreads();
  float* fn = fbuf + n * 2048;
  for (int i = threadIdx.x; i < 2048; i += 256) atomicAdd(&fn[i], fAcc[i]);
}

// ---------------- pass 2: out = relu(Wo'(f^T Qn) + bo') + V0 ----------------
__global__ __launch_bounds__(256) void fmav_pass2(
    const float* __restrict__ x, const _Float16* __restrict__ wf,
    const float* __restrict__ bias, const float* __restrict__ fbuf,
    float* __restrict__ out)
{
  const int n    = blockIdx.y;
  const int wid  = threadIdx.x >> 5;
  const int lane = threadIdx.x & 31;
  const int li   = lane & 15;
  const int hi   = (lane >> 4) & 1;

  __shared__ _Float16 ldsQ[8][16 * 34];   // [pos][qk]
  __shared__ _Float16 ldsY[8][16 * 66];   // [pos][ch]

  const float*    xn  = x + (size_t)n * CHW;
  float*          on  = out + (size_t)n * CHW;
  const _Float16* wqA = wf;
  const _Float16* wvA = wf + 4096;
  const _Float16* woA = wf + 8192;
  const float*    bqF = bias;
  const float*    bvF = bias + 64;
  const float*    boF = bias + 128;

  // f^T as 4 A-fragments (held across the loop)
  v16h fT[4];
  const float* fn = fbuf + n * 2048;
  #pragma unroll
  for (int mb = 0; mb < 4; ++mb) {
    FragCvt fc;
    #pragma unroll
    for (int h = 0; h < 16; ++h) {
      int qk = kA_of(lane, h);
      fc.h16[h] = (_Float16)fn[qk * 64 + mb * 16 + li];
    }
    fT[mb] = fc.h16;
  }

  v8f cq[2], cv[4], co[4];
  #pragma unroll
  for (int mb = 0; mb < 2; ++mb)
    #pragma unroll
    for (int r = 0; r < 8; ++r) cq[mb][r] = bqF[mb * 16 + r + hi * 8];
  #pragma unroll
  for (int mb = 0; mb < 4; ++mb)
    #pragma unroll
    for (int r = 0; r < 8; ++r) { cv[mb][r] = bvF[mb * 16 + r + hi * 8]; co[mb][r] = boF[mb * 16 + r + hi * 8]; }

  const int pBase = (blockIdx.x * 8 + wid) * 64;   // 128 blocks x 8 waves, 4 iters x 16 pos
  _Float16* lq = ldsQ[wid];
  _Float16* ly = ldsY[wid];

  for (int it = 0; it < 4; ++it) {
    const int p0 = pBase + it * 16;
    if (it < 3) __builtin_prefetch(xn + p0 + 16 + (size_t)(hi * 16) * HW, 0, 0);

    const float* xp = xn + p0;
    FragCvt bx0, bx1;
    #pragma unroll
    for (int h = 0; h < 16; ++h) {
      int cA = h + hi * 16;
      bx0.h16[h] = (_Float16)xp[(size_t)cA * HW + li];
      bx1.h16[h] = (_Float16)xp[(size_t)(cA + 32) * HW + li];
    }
    // ---- Q projection + per-position L2 norm -> ldsQ ----
    v8f qd[2];
    #pragma unroll
    for (int mb = 0; mb < 2; ++mb) {
      v16h a0 = *(const v16h*)(wqA + (mb * 2 + 0) * 512 + lane * 16);
      v16h a1 = *(const v16h*)(wqA + (mb * 2 + 1) * 512 + lane * 16);
      v8f acc = cq[mb];
      acc = wmma16(a0, bx0.h16, acc);
      acc = wmma16(a1, bx1.h16, acc);
      qd[mb] = acc;
    }
    float ss = 0.0f;
    #pragma unroll
    for (int mb = 0; mb < 2; ++mb)
      #pragma unroll
      for (int r = 0; r < 8; ++r) ss += qd[mb][r] * qd[mb][r];
    ss += __shfl_xor(ss, 16, 32);
    float rn = 1.0f / fmaxf(sqrtf(ss), NRM_EPS);
    #pragma unroll
    for (int mb = 0; mb < 2; ++mb)
      #pragma unroll
      for (int r = 0; r < 8; ++r)
        lq[li * 34 + mb * 16 + r + hi * 8] = (_Float16)(qd[mb][r] * rn);
    // ---- V0 projection (pre-ReLU, kept for residual) ----
    v8f vd[4];
    #pragma unroll
    for (int mb = 0; mb < 4; ++mb) {
      v16h a0 = *(const v16h*)(wvA + (mb * 2 + 0) * 512 + lane * 16);
      v16h a1 = *(const v16h*)(wvA + (mb * 2 + 1) * 512 + lane * 16);
      v8f acc = cv[mb];
      acc = wmma16(a0, bx0.h16, acc);
      acc = wmma16(a1, bx1.h16, acc);
      vd[mb] = acc;
    }
    // ---- y = f^T (64x32) * Qn (32x16) ----
    FragCvt qc;
    #pragma unroll
    for (int i = 0; i < 8; ++i)
      qc.u[i] = *(const unsigned int*)&lq[li * 34 + 2 * i + hi * 16];
    v8f zc;
    #pragma unroll
    for (int r = 0; r < 8; ++r) zc[r] = 0.0f;
    #pragma unroll
    for (int mb = 0; mb < 4; ++mb) {
      v8f yd = wmma16(fT[mb], qc.h16, zc);
      #pragma unroll
      for (int r = 0; r < 8; ++r)
        ly[li * 66 + mb * 16 + r + hi * 8] = (_Float16)yd[r];
    }
    // ---- out = relu(Wo' y + bo') + V0 ----
    FragCvt y0, y1;
    #pragma unroll
    for (int i = 0; i < 8; ++i) {
      y0.u[i] = *(const unsigned int*)&ly[li * 66 + 2 * i + hi * 16];
      y1.u[i] = *(const unsigned int*)&ly[li * 66 + 32 + 2 * i + hi * 16];
    }
    #pragma unroll
    for (int mb = 0; mb < 4; ++mb) {
      v16h a0 = *(const v16h*)(woA + (mb * 2 + 0) * 512 + lane * 16);
      v16h a1 = *(const v16h*)(woA + (mb * 2 + 1) * 512 + lane * 16);
      v8f od = co[mb];
      od = wmma16(a0, y0.h16, od);
      od = wmma16(a1, y1.h16, od);
      #pragma unroll
      for (int r = 0; r < 8; ++r) {
        float v = fmaxf(od[r], 0.0f) + vd[mb][r];
        on[(size_t)(mb * 16 + r + hi * 8) * HW + p0 + li] = v;
      }
    }
  }
}

extern "C" void kernel_launch(void* const* d_in, const int* in_sizes, int n_in,
                              void* d_out, int out_size, void* d_ws, size_t ws_size,
                              hipStream_t stream) {
  (void)in_sizes; (void)n_in; (void)out_size; (void)ws_size;
  const float* x  = (const float*)d_in[0];
  const float* Wq = (const float*)d_in[1];  const float* bq = (const float*)d_in[2];
  const float* qg = (const float*)d_in[3];  const float* qb = (const float*)d_in[4];
  const float* qm = (const float*)d_in[5];  const float* qv = (const float*)d_in[6];
  const float* Wk = (const float*)d_in[7];  const float* bk = (const float*)d_in[8];
  const float* kg = (const float*)d_in[9];  const float* kb = (const float*)d_in[10];
  const float* km = (const float*)d_in[11]; const float* kv = (const float*)d_in[12];
  const float* Wv = (const float*)d_in[13]; const float* bv = (const float*)d_in[14];
  const float* vg = (const float*)d_in[15]; const float* vb = (const float*)d_in[16];
  const float* vm = (const float*)d_in[17]; const float* vv = (const float*)d_in[18];
  const float* Wo = (const float*)d_in[19]; const float* bo = (const float*)d_in[20];
  const float* og = (const float*)d_in[21]; const float* ob = (const float*)d_in[22];
  const float* om = (const float*)d_in[23]; const float* ov = (const float*)d_in[24];

  float*     fbuf = (float*)((char*)d_ws + F_OFF);
  float*     bias = (float*)((char*)d_ws + B_OFF);
  _Float16*  wf   = (_Float16*)((char*)d_ws + WF_OFF);

  fmav_prep<<<1, 256, 0, stream>>>(Wq, bq, qg, qb, qm, qv,
                                   Wk, bk, kg, kb, km, kv,
                                   Wv, bv, vg, vb, vm, vv,
                                   Wo, bo, og, ob, om, ov,
                                   bias, wf);
  hipMemsetAsync(fbuf, 0, (size_t)NBATCH * 2048 * sizeof(float), stream);
  fmav_pass1<<<dim3(64, NBATCH), 256, 0, stream>>>(x, wf, bias, fbuf);
  fmav_pass2<<<dim3(128, NBATCH), 256, 0, stream>>>(x, wf, bias, fbuf, (float*)d_out);
}
